// RamanAmplifier_67018669687415
// MI455X (gfx1250) — matrix operand order repaired and checked by reading, not compiled, not run on hardware
//
#include <hip/hip_runtime.h>

typedef __attribute__((ext_vector_type(16))) _Float16 v16h;
typedef __attribute__((ext_vector_type(8)))  float    v8f;

#define NUM_PUMPS    4
#define NUM_CHANNELS 100
#define NTOT         104            // pumps + channels
#define NPAD         128            // padded K dim (4 k-tiles of 32)
#define STEPS        500
#define RESP_LEN     801

#define C0F          2.99792458e8f
#define INV_DF       2.0e-11f       // 1 / (80e12/1600)
#define INV_AEFF     1.25e10f       // 1 / 8e-11
#define LN10_OVER_10 0.23025850929940458f
#define LOSS_C0      0.0f
#define LOSS_C1      0.0f
#define LOSS_C2      0.0002f

__global__ __launch_bounds__(256) void raman_rk4_wmma(
    const float* __restrict__ x,        // [B,8] : 4 pump wl, 4 pump power
    const float* __restrict__ raman,    // [801]
    const float* __restrict__ sigwl,    // [100]
    float* __restrict__ out)            // [B,100]
{
    __shared__ float sRaman[RESP_LEN + 3];
    __shared__ float sFreq[NPAD];
    __shared__ float sY[NPAD];                        // G * S result
    __shared__ __align__(32) _Float16 sSh[NPAD];      // stage vector, f16 high part
    __shared__ __align__(32) _Float16 sSl[NPAD];      // stage vector, f16 low  part

    const int tid    = threadIdx.x;
    const int b      = blockIdx.x;
    const int wave   = tid >> 5;        // 8 waves; wave w owns rows [16w,16w+16)
    const int g      = (tid >> 4) & 1;  // lane half-group
    const int lane15 = tid & 15;

    // ---- async copy of Raman response into LDS (CDNA5 async-to-LDS path) ----
    for (int i = tid; i < RESP_LEN; i += 256) {
        unsigned ldsoff = (unsigned)(size_t)(&sRaman[i]);   // low 32 bits = LDS offset
        unsigned goff   = (unsigned)(i * 4);
        asm volatile("global_load_async_to_lds_b32 %0, %1, %2"
                     :: "v"(ldsoff), "v"(goff), "s"(raman) : "memory");
    }

    // ---- per-element initial state / loss / frequency ----
    float sreg = 0.0f, pcur = 0.0f, lossv = 0.0f;
    if (tid < NPAD) {
        float lam = 1.0f, p0 = 0.0f;
        if (tid < NUM_PUMPS) {
            lam = x[b * 8 + tid];
            p0  = fabsf(x[b * 8 + NUM_PUMPS + tid]);   // relu(abs(pw))
        } else if (tid < NTOT) {
            lam = sigwl[tid - NUM_PUMPS];
            p0  = 1.0e-3f;
        }
        sFreq[tid] = (tid < NTOT) ? (C0F / lam) : 1.0f;
        _Float16 hi = (_Float16)p0;
        sSh[tid] = hi;
        sSl[tid] = (_Float16)(p0 - (float)hi);
        sreg = p0;
        pcur = p0;
        lossv = (tid < NTOT)
              ? ((LOSS_C2 + LOSS_C1 * lam + LOSS_C0 * lam * lam) * LN10_OVER_10)
              : 0.0f;
    }
    asm volatile("s_wait_asynccnt 0" ::: "memory");
    __syncthreads();

    // ---- build A-operand (G matrix) tiles directly in WMMA register layout ----
    // A lane layout (16-bit, 16x32): lane = (g*16 + m%16), element h -> K = 16g + h (contiguous order, same as B)
    v16h a_hi[4], a_lo[4];
    {
        const int  row   = wave * 16 + lane15;
        const bool rowok = row < NTOT;
        const float fi   = rowok ? sFreq[row] : 1.0f;
#pragma unroll
        for (int t = 0; t < 4; ++t) {
#pragma unroll
            for (int h = 0; h < 16; ++h) {
                const int k = 32 * t + 16 * g + h;
                float Gv = 0.0f;
                if (rowok && k < NTOT) {
                    float fj   = sFreq[k];
                    float Dv   = fj - fi;                 // D[i,j] = f_j - f_i
                    float fidx = fabsf(Dv) * INV_DF;
                    int   i0   = (int)fidx;
                    if (i0 > RESP_LEN - 2) i0 = RESP_LEN - 2;
                    float w    = fidx - (float)i0;
                    float gr   = sRaman[i0] * (1.0f - w) + sRaman[i0 + 1] * w;
                    gr         = (Dv < 0.0f) ? -gr : gr;
                    float ratio = fi / fj;
                    Gv = gr * fmaxf(1.0f, ratio) * INV_AEFF;
                }
                _Float16 hi = (_Float16)Gv;
                a_hi[t][h] = hi;
                a_lo[t][h] = (_Float16)(Gv - (float)hi);
            }
        }
    }

    // ---- RK4 main loop ----
    const float dz    = 50000.0f / (float)(STEPS - 1);
    const float cS_[3] = {0.5f * dz, 0.5f * dz, dz};
    const float cA_[4] = {dz / 6.0f, dz / 3.0f, dz / 3.0f, dz / 6.0f};

    for (int step = 0; step < STEPS - 1; ++step) {
        float pbase = pcur;
        float acc   = pcur;
#pragma unroll
        for (int s = 0; s < 4; ++s) {
            // y = G * S via split-f16 WMMA; S (pre-split to f16 hi/lo in LDS)
            // replicated across all 16 B columns. B lane-group g holds
            // K = 16g .. 16g+15 of each 32-wide k-tile -> contiguous 32B load.
            v8f c = {};
#pragma unroll
            for (int t = 0; t < 4; ++t) {
                const int base = 32 * t + 16 * g;
                v16h bh = *(const v16h*)(sSh + base);
                v16h bl = *(const v16h*)(sSl + base);
                c = __builtin_amdgcn_wmma_f32_16x16x32_f16(false, a_hi[t], false, bh, (short)0, c, false, false);
                c = __builtin_amdgcn_wmma_f32_16x16x32_f16(false, a_hi[t], false, bl, (short)0, c, false, false);
                c = __builtin_amdgcn_wmma_f32_16x16x32_f16(false, a_lo[t], false, bh, (short)0, c, false, false);
            }
            // all 16 columns of D are identical; lanes (n==0) publish their 8 rows
            if (lane15 == 0) {
                const int rb = wave * 16 + 8 * g;
#pragma unroll
                for (int r = 0; r < 8; ++r) sY[rb + r] = c[r];
            }
            __syncthreads();

            // elementwise RK4 update; each of the 128 update threads owns one
            // element (value kept in sreg) and publishes its f16 hi/lo split.
            if (tid < NPAD) {
                float k = (sY[tid] - lossv) * sreg;
                acc += cA_[s] * k;
                float ns = (s < 3) ? (pbase + cS_[s] * k) : acc;
                sreg = ns;
                _Float16 hi = (_Float16)ns;
                sSh[tid] = hi;
                sSl[tid] = (_Float16)(ns - (float)hi);
                if (s == 3) pcur = acc;
            }
            __syncthreads();
        }
    }

    // ---- emit signal-channel powers ----
    if (tid >= NUM_PUMPS && tid < NTOT)
        out[(size_t)b * NUM_CHANNELS + (tid - NUM_PUMPS)] = pcur;
}

extern "C" void kernel_launch(void* const* d_in, const int* in_sizes, int n_in,
                              void* d_out, int out_size, void* d_ws, size_t ws_size,
                              hipStream_t stream) {
    const float* x     = (const float*)d_in[0];   // [B,8]
    const float* raman = (const float*)d_in[1];   // [801]
    const float* swl   = (const float*)d_in[2];   // [100]
    float*       out   = (float*)d_out;           // [B,100]

    const int B = in_sizes[0] / 8;                // 4096
    raman_rk4_wmma<<<dim3(B), dim3(256), 0, stream>>>(x, raman, swl, out);
}
